// VectorQuantizer_56796647523083
// MI455X (gfx1250) — matrix operand order repaired and checked by reading, not compiled, not run on hardware
//
#include <hip/hip_runtime.h>

// ---------------------------------------------------------------------------
// VQ-VAE vector quantizer for MI455X (gfx1250, wave32, WMMA).
//
// Shapes: z_e [16,64,64,64] f32, emb [1024,64] f32.
// Outputs (concatenated in d_out, f32):
//   [0 .. 4194303]  z_q (straight-through forward value), NCHW
//   [4194304]       vq_loss = 1.25 * mean((z_q - z_e)^2)
//   [4194305 .. ]   indices (b,h,w) as float
//
// Roofline: 8.6 GFLOP matmul vs ~32.5 MB HBM traffic -> memory bound at
// 23.3 TB/s (~1.5us) only if the matmul runs on v_wmma_f32_16x16x32_f16.
// The 32-tile loop software-pipelines LDS B-fragment loads against WMMAs;
// #pragma unroll 2 unrolls it by two (not fully); the compiler emits partial
// s_wait_dscnt waits (verified in round-3 disasm). amdgpu_waves_per_eu(2)
// guarantees the extended-VGPR allocation still fits 2 waves/SIMD so the
// 8-wave workgroup (pinned to 1 block/WGP by 157KB LDS) remains launchable.
// ---------------------------------------------------------------------------

#define KCODES   1024
#define CLAMP_V  1.0e6f
#define ZQ_ELEMS 4194304          // 16*64*64*64

typedef _Float16 v16h __attribute__((ext_vector_type(16)));
typedef float    v8f  __attribute__((ext_vector_type(8)));

union V16H { v16h v; unsigned long long q[4]; };
union H4   { _Float16 h[4]; unsigned long long q; };

// ---- dynamic LDS layout (bytes) -------------------------------------------
// [0, 139264)          codebook f16, 1024 rows x 68 halves (34-dw stride ->
//                      conflict-free 16-lane B-fragment reads, 8B aligned)
// [139264, +17664)     region B, time-multiplexed:
//    phase1: sZ   64x68 halves (8704)     A-fragment staging
//            sEsq 1024 f32     (4096)     min(|emb_k|^2, CLAMP)
//            sRedV/sRedI 128+128 (1024)   cross-wave argmin
//    phase2: sQ   64x69 f32    (17664)    gathered fp32 codebook rows
// [156928, +256)       sIdx 64 int
// [157184, +32)        sLoss 8 f32
#define OFF_CB   0
#define CB_ROW   68
#define OFF_RB   139264
#define RB_SZ    0
#define ZROW     68
#define RB_ESQ   8704
#define RB_REDV  12800
#define RB_REDI  13312
#define SQ_ROW   69
#define OFF_IDX  (OFF_RB + 17664)
#define OFF_LOSS (OFF_IDX + 256)
#define SMEM_BYTES (OFF_LOSS + 32)   // 157216 bytes

// ---- prep: |emb_k|^2 once for the whole grid ------------------------------
__global__ void vq_prep_kernel(const float* __restrict__ emb,
                               float* __restrict__ ws)
{
    int k = blockIdx.x * 256 + threadIdx.x;   // 1024 threads total
    if (k < KCODES) {
        const float4* row = (const float4*)(emb + k * 64);
        float s = 0.0f;
        #pragma unroll
        for (int j = 0; j < 16; ++j) {
            float4 v = row[j];
            s += v.x * v.x + v.y * v.y + v.z * v.z + v.w * v.w;
        }
        ws[1 + k] = s;
    }
}

__global__ __launch_bounds__(256, 1)
__attribute__((amdgpu_waves_per_eu(2)))
void vq_main_kernel(const float* __restrict__ z_e,
                    const float* __restrict__ emb,
                    float* __restrict__ out,
                    float* __restrict__ loss_ws)
{
    extern __shared__ char smem[];
    _Float16* sCB   = (_Float16*)(smem + OFF_CB);
    _Float16* sZ    = (_Float16*)(smem + OFF_RB + RB_SZ);
    float*    sEsq  = (float*)   (smem + OFF_RB + RB_ESQ);
    float*    sRedV = (float*)   (smem + OFF_RB + RB_REDV);
    int*      sRedI = (int*)     (smem + OFF_RB + RB_REDI);
    float*    sQ    = (float*)   (smem + OFF_RB);           // phase2 overlay
    int*      sIdx  = (int*)     (smem + OFF_IDX);
    float*    sLoss = (float*)   (smem + OFF_LOSS);

    const int t    = threadIdx.x;
    const int lane = t & 31;
    const int wave = t >> 5;
    const int bh   = blockIdx.x;          // b*64 + h
    const int b    = bh >> 6;
    const int h    = bh & 63;
    const size_t zbase = (size_t)b * 262144u + (size_t)h * 64u;

    // ---- esq -> LDS with clamp pre-applied --------------------------------
    for (int k = t; k < KCODES; k += 256)
        sEsq[k] = fminf(loss_ws[1 + k], CLAMP_V);

    // ---- codebook -> f16 LDS (float4 -> 4 halves, 8B LDS stores) ----------
    {
        const float4* embq = (const float4*)emb;
        for (int i = 0; i < 64; ++i) {
            int e4 = t + 256 * i;         // quad index: k*16 + dq
            int k  = e4 >> 4;
            int d  = (e4 & 15) * 4;
            float4 v = embq[e4];
            H4 hv;
            hv.h[0] = (_Float16)v.x; hv.h[1] = (_Float16)v.y;
            hv.h[2] = (_Float16)v.z; hv.h[3] = (_Float16)v.w;
            *(unsigned long long*)(sCB + k * CB_ROW + d) = hv.q;
        }
    }

    // ---- z tile (64 w-rows x 64 dims) -> f16 LDS (transposed staging) -----
    for (int i = 0; i < 4; ++i) {
        int e4 = t + 256 * i;             // quad index: d*16 + w4
        int d  = e4 >> 4;
        int w4 = (e4 & 15) * 4;
        float4 v = *(const float4*)(z_e + zbase + (size_t)d * 4096u + (size_t)w4);
        sZ[(w4 + 0) * ZROW + d] = (_Float16)v.x;
        sZ[(w4 + 1) * ZROW + d] = (_Float16)v.y;
        sZ[(w4 + 2) * ZROW + d] = (_Float16)v.z;
        sZ[(w4 + 3) * ZROW + d] = (_Float16)v.w;
    }
    __syncthreads();

    // ---- WMMA distance + running argmin -----------------------------------
    const int  mtile = wave & 3;          // which 16 rows of the tile
    const int  half  = wave >> 2;         // which 512 codes
    const int  m0    = mtile * 16;
    const int  lmod  = lane & 15;
    const bool hi    = lane >= 16;

    // A fragments (16x32 f16 layout, ISA 7.12.2):
    //   lanes 0-15:  M=lane,    halves = K{0..7} then K{16..23}
    //   lanes 16-31: M=lane-16, halves = K{8..15} then K{24..31}
    V16H a0, a1;
    {
        const int row  = m0 + lmod;
        const int aoff = hi ? 8 : 0;
        const unsigned long long* zp =
            (const unsigned long long*)(sZ + row * ZROW);   // 8B aligned
        a0.q[0] = zp[(aoff +  0) >> 2];  a0.q[1] = zp[(aoff +  4) >> 2];
        a0.q[2] = zp[(aoff + 16) >> 2];  a0.q[3] = zp[(aoff + 20) >> 2];
        a1.q[0] = zp[(aoff + 32) >> 2];  a1.q[1] = zp[(aoff + 36) >> 2];
        a1.q[2] = zp[(aoff + 48) >> 2];  a1.q[3] = zp[(aoff + 52) >> 2];
    }

    float best[8];
    int   bidx[8];
    #pragma unroll
    for (int r = 0; r < 8; ++r) { best[r] = 3.0e38f; bidx[r] = 0; }

    const int koff2 = hi ? 2 : 0;         // B frag qword offset (8 halves)
    const int ntBeg = half * 32;

    auto loadB = [&](int nt, V16H& bb0, V16H& bb1, float& esq) {
        const int code = nt * 16 + lmod;
        const unsigned long long* bp =
            (const unsigned long long*)(sCB + code * CB_ROW);
        bb0.q[0] = bp[koff2 + 0];  bb0.q[1] = bp[koff2 + 1];
        bb0.q[2] = bp[koff2 + 4];  bb0.q[3] = bp[koff2 + 5];
        bb1.q[0] = bp[koff2 + 8];  bb1.q[1] = bp[koff2 + 9];
        bb1.q[2] = bp[koff2 + 12]; bb1.q[3] = bp[koff2 + 13];
        esq = sEsq[code];
    };

    auto epilogue = [&](const v8f& acc, float esq, int code) {
        // dist = min(|e|^2,C) - 2*clip(dot,+-C)  (|z|^2 row-constant: dropped)
        #pragma unroll
        for (int r = 0; r < 8; ++r) {
            float dot  = fminf(fmaxf(acc[r], -CLAMP_V), CLAMP_V);
            float dist = esq - 2.0f * dot;
            if (dist < best[r]) { best[r] = dist; bidx[r] = code; }
        }
    };

    // pipeline prologue: load tile0, issue its WMMAs while loading tile1
    V16H b0c, b1c; float esqc; int codec;
    loadB(ntBeg, b0c, b1c, esqc);
    codec = ntBeg * 16 + lmod;

    v8f accP;                              // accumulator awaiting epilogue
    float esqP; int codeP;
    {
        V16H b0n, b1n; float esqn;
        loadB(ntBeg + 1, b0n, b1n, esqn);
        v8f acc = {};
        acc = __builtin_amdgcn_wmma_f32_16x16x32_f16(
                  false, a0.v, false, b0c.v, (short)0, acc, false, false);
        acc = __builtin_amdgcn_wmma_f32_16x16x32_f16(
                  false, a1.v, false, b1c.v, (short)0, acc, false, false);
        accP = acc; esqP = esqc; codeP = codec;
        b0c = b0n; b1c = b1n; esqc = esqn; codec = (ntBeg + 1) * 16 + lmod;
    }

    #pragma unroll 2
    for (int i = 1; i < 32; ++i) {         // constant trip count -> scalar loop
        const int nt  = ntBeg + i;
        const int ntn = (i < 31) ? nt + 1 : nt;

        // stage A: prefetch next tile's B fragments + esq
        V16H b0n, b1n; float esqn;
        loadB(ntn, b0n, b1n, esqn);

        // stage B: WMMAs for the current tile
        v8f acc = {};
        acc = __builtin_amdgcn_wmma_f32_16x16x32_f16(
                  false, a0.v, false, b0c.v, (short)0, acc, false, false);
        acc = __builtin_amdgcn_wmma_f32_16x16x32_f16(
                  false, a1.v, false, b1c.v, (short)0, acc, false, false);

        // stage C: argmin epilogue for the PREVIOUS tile (hazard-free)
        epilogue(accP, esqP, codeP);

        accP = acc; esqP = esqc; codeP = codec;
        b0c = b0n; b1c = b1n; esqc = esqn; codec = ntn * 16 + lmod;
    }
    epilogue(accP, esqP, codeP);           // drain the pipeline

    // butterfly argmin across the 16 lanes sharing each row set (tie->low idx)
    #pragma unroll
    for (int s = 1; s < 16; s <<= 1) {
        #pragma unroll
        for (int r = 0; r < 8; ++r) {
            float ov = __shfl_xor(best[r], s, 32);
            int   oi = __shfl_xor(bidx[r], s, 32);
            if (ov < best[r] || (ov == best[r] && oi < bidx[r])) {
                best[r] = ov; bidx[r] = oi;
            }
        }
    }
    if (lane == 0 || lane == 16) {
        const int rbase = half * 64 + m0 + (hi ? 8 : 0);
        #pragma unroll
        for (int r = 0; r < 8; ++r) {
            sRedV[rbase + r] = best[r];
            sRedI[rbase + r] = bidx[r];
        }
    }
    __syncthreads();

    // combine the two 512-code halves; emit indices (codes<512 win ties)
    if (t < 64) {
        float v0 = sRedV[t];      int i0 = sRedI[t];
        float v1 = sRedV[64 + t]; int i1 = sRedI[64 + t];
        int sel = (v1 < v0) ? i1 : i0;
        sIdx[t] = sel;
        out[(size_t)ZQ_ELEMS + 1u + (size_t)bh * 64u + (size_t)t] = (float)sel;
    }
    __syncthreads();

    // ---- gather selected fp32 codebook rows (coalesced float4 reads) ------
    for (int i = 0; i < 4; ++i) {
        int e4 = t + 256 * i;             // quad index: w*16 + dq
        int w  = e4 >> 4;
        int dq = e4 & 15;
        float4 v = ((const float4*)(emb + sIdx[w] * 64))[dq];
        int base = w * SQ_ROW + dq * 4;
        sQ[base + 0] = v.x; sQ[base + 1] = v.y;
        sQ[base + 2] = v.z; sQ[base + 3] = v.w;
    }
    __syncthreads();

    // ---- z_q NCHW store + fused 1.25*MSE loss (float4 global traffic) -----
    float lsum = 0.0f;
    for (int i = 0; i < 4; ++i) {
        int e4 = t + 256 * i;             // quad index: d*16 + w4
        int d  = e4 >> 4;
        int w4 = (e4 & 15) * 4;
        float q0 = sQ[(w4 + 0) * SQ_ROW + d];
        float q1 = sQ[(w4 + 1) * SQ_ROW + d];
        float q2 = sQ[(w4 + 2) * SQ_ROW + d];
        float q3 = sQ[(w4 + 3) * SQ_ROW + d];
        size_t g = zbase + (size_t)d * 4096u + (size_t)w4;
        float4 ze = *(const float4*)(z_e + g);
        float4 qv; qv.x = q0; qv.y = q1; qv.z = q2; qv.w = q3;
        *(float4*)(out + g) = qv;
        float d0 = q0 - ze.x, d1 = q1 - ze.y, d2 = q2 - ze.z, d3 = q3 - ze.w;
        lsum += d0 * d0 + d1 * d1 + d2 * d2 + d3 * d3;
    }
    #pragma unroll
    for (int m = 16; m > 0; m >>= 1) lsum += __shfl_xor(lsum, m, 32);
    if (lane == 0) sLoss[wave] = lsum;
    __syncthreads();
    if (t == 0) {
        float s = 0.0f;
        #pragma unroll
        for (int wv = 0; wv < 8; ++wv) s += sLoss[wv];
        atomicAdd(loss_ws, s);
    }
}

__global__ void vq_finalize_kernel(const float* __restrict__ ws,
                                   float* __restrict__ out)
{
    // vq_loss = (1 + 0.25) * sum / (16*64*64*64)
    out[ZQ_ELEMS] = ws[0] * (1.25f / 4194304.0f);
}

extern "C" void kernel_launch(void* const* d_in, const int* in_sizes, int n_in,
                              void* d_out, int out_size, void* d_ws, size_t ws_size,
                              hipStream_t stream)
{
    const float* z_e = (const float*)d_in[0];   // [16,64,64,64]
    const float* emb = (const float*)d_in[1];   // [1024,64]
    float* out = (float*)d_out;
    float* ws  = (float*)d_ws;

    // zero the loss accumulator every launch (graph-capture legal)
    hipMemsetAsync(d_ws, 0, sizeof(float), stream);

    // ws[1..1024] = |emb_k|^2
    vq_prep_kernel<<<4, 256, 0, stream>>>(emb, ws);

    // one block per (b,h) pair: 16*64 = 1024 blocks, 8 waves each
    vq_main_kernel<<<1024, 256, SMEM_BYTES, stream>>>(z_e, emb, out, ws);
    vq_finalize_kernel<<<1, 1, 0, stream>>>(ws, out);
}
